// PhasorSample_88914412961875
// MI455X (gfx1250) — compile-verified, matrix operand order checked by reading, not run on hardware
//
#include <hip/hip_runtime.h>
#include <hip/hip_bf16.h>

typedef __attribute__((ext_vector_type(16))) _Float16 v16h;
typedef __attribute__((ext_vector_type(8)))  float    v8f;

#define NPTS            262144
#define WAVES_PER_BLOCK 8
#define PTS_PER_BLOCK   (WAVES_PER_BLOCK * 16)

// ---------------------------------------------------------------------------
// Prep kernel: build B matrix [64 k-chunks][ri(2)][step(4)][lane(32)][v(8)]
// as packed f16 pairs (dwords) in the wave32 WMMA-B register layout.
//   B column (k,ri,c); B row kk in 0..127:  kk<64 -> cy coefficient row l=kk,
//   kk>=64 -> sy coefficient row l=kk-64.
//   ri=0 (Tr): [ Kr*g*w ; -Ki*g*w ]   ri=1 (Ti): [ Ki*g*w ; Kr*g*w ]
// WMMA-B element mapping (mirror of documented 16-bit A layout):
//   lane = c + 16*kk[3], vgpr v = (kk[4]<<2)|(kk[2:1]), half = kk[0]
// ---------------------------------------------------------------------------
__global__ __launch_bounds__(256) void phasor_prep(
    const float* __restrict__ pr, const float* __restrict__ pim,
    const float* __restrict__ gvar, unsigned* __restrict__ Bpk)
{
    int gid  = blockIdx.x * 256 + threadIdx.x;      // 0 .. 131071
    int k    = gid >> 11;
    int r2   = gid & 2047;
    int ri   = r2 >> 10;
    int r3   = r2 & 1023;
    int s    = r3 >> 8;
    int lane = (r3 >> 3) & 31;
    int v    = r3 & 7;
    int c    = lane & 15;

    float var = gvar[0];
    int   kx  = (k < 32) ? k : k - 64;
    float fx  = (float)kx * (1.0f / 512.0f);
    const float pi2 = 9.8696044010893586f;          // pi^2

    unsigned dw = 0;
#pragma unroll
    for (int h = 0; h < 2; ++h) {
        int kk   = s * 32 + ((v >> 2) << 4) + ((lane >> 4) << 3) + ((v & 3) << 1) + h;
        int l    = kk & 63;
        int part = kk >> 6;                         // 0: cy row, 1: sy row
        float fy  = (float)l * (1.0f / 512.0f);
        float g   = __expf(-2.0f * pi2 * var * (fx * fx + fy * fy));
        float wgt = (l == 0) ? 1.0f : 2.0f;
        float Kr  = pr [c * 4096 + k * 64 + l] * g * wgt;
        float Ki  = pim[c * 4096 + k * 64 + l] * g * wgt;
        float val = (ri == 0) ? ((part == 0) ?  Kr : -Ki)
                              : ((part == 0) ?  Ki :  Kr);
        _Float16 hv = (_Float16)val;
        dw |= ((unsigned)__builtin_bit_cast(unsigned short, hv)) << (16 * h);
    }
    Bpk[gid] = dw;
}

// ---------------------------------------------------------------------------
// Main kernel: 8 waves/block, 16 points/wave.
//   A (registers): [cy(2pi*y*l) | sy(2pi*y*l)] f16, K=128 -> 4 x v16h
//   per k: 2 WMMA accumulators (Tr, Ti) over K=128 (4 wmma each),
//   fused epilogue feat += cx*Tr - sx*Ti from per-wave LDS twiddle table.
//   B chunks double-buffered into LDS via global_load_async_to_lds_b128.
// ---------------------------------------------------------------------------
__global__ __launch_bounds__(256) void phasor_main(
    const float* __restrict__ coord, const unsigned* __restrict__ Bpk,
    float* __restrict__ out)
{
    __shared__ __attribute__((aligned(32))) unsigned ldsB[2][2048]; // 2 x 8KB
    __shared__ float csTab[WAVES_PER_BLOCK][2][16][64];             // 64KB

    const int tid  = threadIdx.x;
    const int lane = tid & 31;
    const int wave = tid >> 5;
    const int hi   = lane >> 4;
    const int m    = lane & 15;
    const int tileBase = blockIdx.x * PTS_PER_BLOCK + wave * 16;

    // kick off async fill of buffer 0 with B chunk k=0 (overlaps trig below)
    {
        unsigned ldsOff = (unsigned)(size_t)&ldsB[0][0] + (unsigned)tid * 16u;
        unsigned goff   = (unsigned)tid * 16u;
        const unsigned char* gbase = (const unsigned char*)Bpk;
        asm volatile("global_load_async_to_lds_b128 %0, %1, %2"
                     :: "v"(ldsOff), "v"(goff), "s"(gbase) : "memory");
        asm volatile("global_load_async_to_lds_b128 %0, %1, %2"
                     :: "v"(ldsOff + 4096u), "v"(goff + 4096u), "s"(gbase) : "memory");
    }

    const float x = coord[(tileBase + m) * 2 + 0];
    const float y = coord[(tileBase + m) * 2 + 1];

    // stage-2 twiddles: csTab[wave][0][m][k]=cos(2pi*x*kx), [1]=sin
    // (v_cos/v_sin take revolutions: arg = x*kx directly)
#pragma unroll 4
    for (int j = 0; j < 32; ++j) {
        int   k  = hi * 32 + j;
        float kx = (k < 32) ? (float)k : (float)(k - 64);
        float t  = x * kx;
        csTab[wave][0][m][k] = __builtin_amdgcn_cosf(t);
        csTab[wave][1][m][k] = __builtin_amdgcn_sinf(t);
    }

    // A operand in WMMA-A layout: lane = m + 16*kk[3]; v16h elem e=(v<<1)|h
    v16h A[4];
#pragma unroll
    for (int s = 0; s < 4; ++s) {
#pragma unroll
        for (int e = 0; e < 16; ++e) {
            int v  = e >> 1, h = e & 1;
            int kk = s * 32 + ((v >> 2) << 4) + (hi << 3) + ((v & 3) << 1) + h;
            int l  = kk & 63;
            float t   = y * (float)l;
            float val = (kk < 64) ? __builtin_amdgcn_cosf(t)
                                  : __builtin_amdgcn_sinf(t);
            A[s][e] = (_Float16)val;
        }
    }

    v8f accF = {0.f, 0.f, 0.f, 0.f, 0.f, 0.f, 0.f, 0.f};

    asm volatile("s_wait_asynccnt 0" ::: "memory");
    __syncthreads();

    int p = 0;
    for (int k = 0; k < 64; ++k) {
        if (k < 63) {   // prefetch next chunk into the other buffer
            unsigned ldsOff = (unsigned)(size_t)&ldsB[p ^ 1][0] + (unsigned)tid * 16u;
            unsigned goff   = (unsigned)tid * 16u;
            const unsigned char* gbase = (const unsigned char*)(Bpk + (k + 1) * 2048);
            asm volatile("global_load_async_to_lds_b128 %0, %1, %2"
                         :: "v"(ldsOff), "v"(goff), "s"(gbase) : "memory");
            asm volatile("global_load_async_to_lds_b128 %0, %1, %2"
                         :: "v"(ldsOff + 4096u), "v"(goff + 4096u), "s"(gbase) : "memory");
        }

        const unsigned* tb = &ldsB[p][0];
        v8f ar = {0.f, 0.f, 0.f, 0.f, 0.f, 0.f, 0.f, 0.f};
        v8f ai = {0.f, 0.f, 0.f, 0.f, 0.f, 0.f, 0.f, 0.f};
#pragma unroll
        for (int s = 0; s < 4; ++s) {
            v16h b0 = *(const v16h*)(tb + 0 * 1024 + s * 256 + lane * 8);
            ar = __builtin_amdgcn_wmma_f32_16x16x32_f16(
                    false, A[s], false, b0, (short)0, ar, false, false);
            v16h b1 = *(const v16h*)(tb + 1 * 1024 + s * 256 + lane * 8);
            ai = __builtin_amdgcn_wmma_f32_16x16x32_f16(
                    false, A[s], false, b1, (short)0, ai, false, false);
        }

        // fused stage 2: feat[row, c=lane&15] += cx*Tr - sx*Ti
#pragma unroll
        for (int v = 0; v < 8; ++v) {
            int   row = v + hi * 8;
            float cx  = csTab[wave][0][row][k];
            float sx  = csTab[wave][1][row][k];
            accF[v] += cx * ar[v] - sx * ai[v];
        }

        asm volatile("s_wait_asynccnt 0" ::: "memory");
        __syncthreads();
        p ^= 1;
    }

#pragma unroll
    for (int v = 0; v < 8; ++v) {
        int row = v + hi * 8;
        out[(size_t)(tileBase + row) * 16 + m] = accF[v];
    }
}

extern "C" void kernel_launch(void* const* d_in, const int* in_sizes, int n_in,
                              void* d_out, int out_size, void* d_ws, size_t ws_size,
                              hipStream_t stream) {
    const float* coord = (const float*)d_in[0];   // [262144,2]
    const float* pr    = (const float*)d_in[1];   // [1,16,64,64]
    const float* pim   = (const float*)d_in[2];   // [1,16,64,64]
    const float* gvar  = (const float*)d_in[3];   // scalar
    float*       out   = (float*)d_out;           // [262144,16]
    unsigned*    Bpk   = (unsigned*)d_ws;         // 512KB packed f16 B matrix

    phasor_prep<<<512, 256, 0, stream>>>(pr, pim, gvar, Bpk);
    phasor_main<<<NPTS / PTS_PER_BLOCK, 256, 0, stream>>>(coord, Bpk, out);
}
